// CGAPB_84052509983238
// MI455X (gfx1250) — compile-verified, matrix operand order checked by reading, not run on hardware
//
#include <hip/hip_runtime.h>
#include <hip/hip_bf16.h>

// ---------------------------------------------------------------------------
// EGNN pool/depool for MI455X (gfx1250, wave32, WMMA bf16 K=32, f32 accum).
//
// Input flattening assumption (JAX pytree order; nested dicts sorted by key):
//  0 x(8192,3) 1 node_feats 2 edge_index(unused; graph static) 3 edge_attr
//  4 distances 5 bead_node_feats 6 bead_edge_index(unused) 7 bead_edge_attr
//  8.. pool_params:  +0 Wout +1 bout +2 edge_emb
//                    +3+10l: Wc We1 We2 Wn1 Wn2 bc be1 be2 bn1 bn2 (l=0,1)
//                    +23 node_emb
//  32.. depool_params (same layout)     56 batch_size
// ---------------------------------------------------------------------------

#define BB 128
#define NN_ 64
#define NBEAD 16
#define NRES 32
#define HH 128

typedef __attribute__((ext_vector_type(16))) __bf16 v16bf;
typedef __attribute__((ext_vector_type(8)))  float  v8f;

__device__ __forceinline__ v8f wmma_bf16(v16bf a, v16bf b, v8f c) {
  return __builtin_amdgcn_wmma_f32_16x16x32_bf16(false, a, false, b, (short)0, c,
                                                 false, false);
}
__device__ __forceinline__ int kmap(int h, int hi) {
  return h + ((h >= 8) ? 8 : 0) + 8 * hi;   // 16-bit A/B fragment K mapping
}
__device__ __forceinline__ float silu_f(float x) { return x / (1.0f + __expf(-x)); }
__device__ __forceinline__ v8f splat8(float v) {
  v8f r;
#pragma unroll
  for (int i = 0; i < 8; ++i) r[i] = v;
  return r;
}
// B fragment from fragment-contiguous prepacked weights (global or LDS):
// one 2x b128 load per lane.
__device__ __forceinline__ v16bf ldfrag(const __bf16* W, int kc, int nt, int NT,
                                        int lane) {
  return *reinterpret_cast<const v16bf*>(W + ((size_t)((kc * NT + nt) * 32 + lane) << 4));
}

// --------------------------- weight prepack --------------------------------
// dst layout: [kc][nt][lane 0..31][16 halves] bf16. we1mode remaps rows so the
// packed K order is [h_r(0:128) h_s(128:256) e(256:384) d2(384) be1(385) pad],
// with the bias folded in as row 385 (activation supplies a constant 1.0).
__global__ void eg_prep_frag(const float* __restrict__ src,
                             const float* __restrict__ bias,
                             __bf16* __restrict__ dst,
                             int Ksrc, int Kpad, int Nn, int we1mode) {
  int NT = Nn >> 4;
  int total = (Kpad >> 5) * NT * 512;
  for (int t = blockIdx.x * blockDim.x + threadIdx.x; t < total;
       t += gridDim.x * blockDim.x) {
    int hh = t & 15, lane = (t >> 4) & 31, fi = t >> 9;
    int nt = fi % NT, kc = fi / NT, hi = lane >> 4;
    int K = kc * 32 + kmap(hh, hi);
    int col = nt * 16 + (lane & 15);
    float v = 0.0f;
    if (we1mode) {
      if (K < 256)       v = src[K * Nn + col];
      else if (K < 384)  v = src[(K + 1) * Nn + col];  // e features (257..384)
      else if (K == 384) v = src[256 * Nn + col];      // d2 row
      else if (K == 385) v = bias[col];                // folded be1
    } else {
      if (K < Ksrc)      v = src[K * Nn + col];
    }
    dst[t] = (__bf16)v;
  }
}
__global__ void eg_prep_bf16(const float* __restrict__ s, __bf16* __restrict__ d, int n) {
  int t = blockIdx.x * blockDim.x + threadIdx.x;
  if (t < n) d[t] = (__bf16)s[t];
}
__global__ void eg_zero_scal(float* s) { if (threadIdx.x < 4) s[threadIdx.x] = 0.0f; }
__global__ void eg_copy_f32(const float* __restrict__ s, float* __restrict__ d, int n) {
  int t = blockIdx.x * blockDim.x + threadIdx.x;
  if (t < n) d[t] = s[t];
}
__global__ void eg_init_h(const float* __restrict__ emb, const int* __restrict__ feats,
                          float* __restrict__ h, int num_nodes) {
  int t = blockIdx.x * blockDim.x + threadIdx.x;
  if (t < num_nodes * 128) h[t] = emb[feats[t >> 7] * 128 + (t & 127)];
}
__global__ void eg_xupd(float* __restrict__ xc, const float* __restrict__ xa,
                        float inv_n, int n3) {
  int t = blockIdx.x * blockDim.x + threadIdx.x;
  if (t < n3) xc[t] += xa[t] * inv_n;
}

// ------------------------------ edge kernel --------------------------------
// One workgroup = (batch b, 16 receivers). 4 waves stride over senders.
// Weights live in LDS (weight-stationary; 320KB/WGP on CDNA5):
//   We1 frags 104KB + We2 frags 32KB, streamed in once per workgroup.
// Per 16-edge tile (fixed sender, 16 receivers):
//   m1 = silu([h_r|h_s|e|d2|1] @ We1')      13 x 8 WMMA, C starts at inline 0
//   m2 = silu(m1 @ We2 + be2)                4 x 8 WMMA
//   w  = tanh(m2 @ Wc + bc)                  cross-lane VALU reduce
// segment_sum over senders == running sum of the WMMA C accumulators.
#define E_WE1 0
#define E_WE2 106496
#define E_M1  139264
#define E_AGG 155648
#define E_WL  163840
#define E_X   167936
#define E_XR  168704
#define E_LDS_BYTES 168896

__global__ void __launch_bounds__(128)
eg_edge(const float* __restrict__ h, const float* __restrict__ xcur,
        const __bf16* __restrict__ eemb, const int* __restrict__ eattr,
        const __bf16* __restrict__ We1f,
        const __bf16* __restrict__ We2f, const float* __restrict__ be2,
        const float* __restrict__ Wc, const float* __restrict__ bc,
        float* __restrict__ agg, float* __restrict__ xagg, int n_per) {
  extern __shared__ __align__(16) char smem[];
  const int lane = threadIdx.x & 31;
  const int wave = threadIdx.x >> 5;
  const int jt = n_per >> 4;
  const int b = blockIdx.x / jt;
  const int j0 = (blockIdx.x - b * jt) * 16;
  const int nb0 = b * n_per;
  const int M = lane & 15;
  const int hi = lane >> 4;

  __bf16* we1s  = (__bf16*)(smem + E_WE1);
  __bf16* we2s  = (__bf16*)(smem + E_WE2);
  __bf16* m1s   = (__bf16*)(smem + E_M1) + wave * 2048;
  float*  aggred= (float*)(smem + E_AGG);
  float*  wl    = (float*)(smem + E_WL) + wave * 256;
  float*  xlds  = (float*)(smem + E_X);
  float*  xred  = (float*)(smem + E_XR);

  // stream weight fragments into LDS (b128 copies)
  {
    const uint4* s1 = (const uint4*)We1f; uint4* d1 = (uint4*)we1s;
    for (int t = threadIdx.x; t < 6656; t += 128) d1[t] = s1[t];
    const uint4* s2 = (const uint4*)We2f; uint4* d2 = (uint4*)we2s;
    for (int t = threadIdx.x; t < 2048; t += 128) d2[t] = s2[t];
  }
  for (int t = threadIdx.x; t < n_per * 3; t += 128) xlds[t] = xcur[nb0 * 3 + t];
  for (int t = threadIdx.x; t < 2048; t += 128) aggred[t] = 0.0f;
  if (threadIdx.x < 48) xred[threadIdx.x] = 0.0f;
  __syncthreads();

  // receiver-side A fragments (h_r) are sender-invariant: keep in registers
  v16bf a_hr[4];
#pragma unroll
  for (int kc = 0; kc < 4; ++kc) {
    const float* hr = h + (size_t)(nb0 + j0 + M) * 128 + kc * 32;
#pragma unroll
    for (int t = 0; t < 16; ++t) a_hr[kc][t] = (__bf16)hr[kmap(t, hi)];
  }

  v8f aggC[8];
#pragma unroll
  for (int nt = 0; nt < 8; ++nt) aggC[nt] = splat8(0.0f);
  float xp0 = 0.f, xp1 = 0.f, xp2 = 0.f;
  const float bcv = bc[0];

  int pin = 0;                                     // anti-LICM pin (stays 0)
  for (int i = wave; i < n_per; i += 4) {          // uniform across waves
    asm volatile("" : "+v"(pin));                  // keep B-frag loads in-loop
    const __bf16* w1p = we1s + pin;
    const __bf16* w2p = we2s + pin;

    const float dx = xlds[i * 3 + 0] - xlds[(j0 + M) * 3 + 0];
    const float dy = xlds[i * 3 + 1] - xlds[(j0 + M) * 3 + 1];
    const float dz = xlds[i * 3 + 2] - xlds[(j0 + M) * 3 + 2];
    const float d2v = dx * dx + dy * dy + dz * dz;
    const __bf16* erow =
        eemb + (size_t)eattr[(size_t)b * n_per * n_per + i * n_per + (j0 + M)] * 128;
    const float* hs = h + (size_t)(nb0 + i) * 128;

    v8f c1[8] = {};                                // zeroinit -> SRC2 inline 0
    for (int kc = 0; kc < 13; ++kc) {
      v16bf a;
      if (kc < 4) {
        a = a_hr[kc];
      } else if (kc < 8) {
#pragma unroll
        for (int t = 0; t < 16; ++t) a[t] = (__bf16)hs[(kc - 4) * 32 + kmap(t, hi)];
      } else if (kc < 12) {
#pragma unroll
        for (int t = 0; t < 16; ++t) a[t] = erow[(kc - 8) * 32 + kmap(t, hi)];
      } else {
#pragma unroll
        for (int t = 0; t < 16; ++t) a[t] = (__bf16)0.0f;
        a[0] = hi ? (__bf16)0.0f : (__bf16)d2v;   // d2  at K=384
        a[1] = hi ? (__bf16)0.0f : (__bf16)1.0f;  // 1.0 at K=385 (bias column)
      }
#pragma unroll
      for (int nt = 0; nt < 8; ++nt)
        c1[nt] = wmma_bf16(a, ldfrag(w1p, kc, nt, 8, lane), c1[nt]);
    }

    __syncthreads();                               // protect staging reuse
#pragma unroll
    for (int nt = 0; nt < 8; ++nt)
#pragma unroll
      for (int v = 0; v < 8; ++v)
        m1s[(v + 8 * hi) * 128 + nt * 16 + M] = (__bf16)silu_f(c1[nt][v]);
    __syncthreads();

    v8f c2[8];
#pragma unroll
    for (int nt = 0; nt < 8; ++nt) c2[nt] = splat8(be2[nt * 16 + M]);
#pragma unroll
    for (int kc = 0; kc < 4; ++kc) {
      v16bf a;
#pragma unroll
      for (int t = 0; t < 16; ++t) a[t] = m1s[M * 128 + kc * 32 + kmap(t, hi)];
#pragma unroll
      for (int nt = 0; nt < 8; ++nt)
        c2[nt] = wmma_bf16(a, ldfrag(w2p, kc, nt, 8, lane), c2[nt]);
    }

    float wpart[8];
#pragma unroll
    for (int v = 0; v < 8; ++v) wpart[v] = 0.0f;
#pragma unroll
    for (int nt = 0; nt < 8; ++nt) {
      const float wc = Wc[nt * 16 + M];
#pragma unroll
      for (int v = 0; v < 8; ++v) {
        const float s = silu_f(c2[nt][v]);
        aggC[nt][v] += s;                          // segment_sum over senders
        wpart[v] += s * wc;
      }
    }
#pragma unroll
    for (int v = 0; v < 8; ++v) wl[(v + 8 * hi) * 16 + M] = wpart[v];
    __syncthreads();
    if (lane < 16) {                               // lane == its receiver row
      float ws = bcv;
#pragma unroll
      for (int n2 = 0; n2 < 16; ++n2) ws += wl[lane * 16 + n2];
      const float w = tanhf(ws);
      xp0 += dx * w; xp1 += dy * w; xp2 += dz * w;
    }
    __syncthreads();
  }

  // cross-wave reduction (LDS float atomics), then exclusive global stores
#pragma unroll
  for (int nt = 0; nt < 8; ++nt)
#pragma unroll
    for (int v = 0; v < 8; ++v)
      atomicAdd(&aggred[(v + 8 * hi) * 128 + nt * 16 + M], aggC[nt][v]);
  if (lane < 16) {
    atomicAdd(&xred[lane * 3 + 0], xp0);
    atomicAdd(&xred[lane * 3 + 1], xp1);
    atomicAdd(&xred[lane * 3 + 2], xp2);
  }
  __syncthreads();
  for (int t = threadIdx.x; t < 2048; t += 128)
    agg[(size_t)(nb0 + j0 + (t >> 7)) * 128 + (t & 127)] = aggred[t];
  if (threadIdx.x < 48) xagg[(nb0 + j0) * 3 + threadIdx.x] = xred[threadIdx.x];
}

// ------------------------------ node kernel --------------------------------
// hu = silu([h|agg] @ Wn1 + bn1) ; h += hu @ Wn2 + bn2.  One wave / 16 nodes.
__global__ void __launch_bounds__(128)
eg_node(float* __restrict__ h, const float* __restrict__ agg,
        const __bf16* __restrict__ Wn1f, const float* __restrict__ bn1,
        const __bf16* __restrict__ Wn2f, const float* __restrict__ bn2) {
  const int lane = threadIdx.x & 31;
  const int wave = threadIdx.x >> 5;
  const int m0 = (blockIdx.x * 4 + wave) * 16;
  const int M = lane & 15, hi = lane >> 4;
  __shared__ __bf16 hulds[4][2048];

  v8f c[8];
#pragma unroll
  for (int nt = 0; nt < 8; ++nt) c[nt] = splat8(bn1[nt * 16 + M]);
#pragma unroll
  for (int kc = 0; kc < 8; ++kc) {
    const float* src = (kc < 4) ? (h + (size_t)(m0 + M) * 128 + kc * 32)
                                : (agg + (size_t)(m0 + M) * 128 + (kc - 4) * 32);
    v16bf a;
#pragma unroll
    for (int t = 0; t < 16; ++t) a[t] = (__bf16)src[kmap(t, hi)];
#pragma unroll
    for (int nt = 0; nt < 8; ++nt)
      c[nt] = wmma_bf16(a, ldfrag(Wn1f, kc, nt, 8, lane), c[nt]);
  }
#pragma unroll
  for (int nt = 0; nt < 8; ++nt)
#pragma unroll
    for (int v = 0; v < 8; ++v)
      hulds[wave][(v + 8 * hi) * 128 + nt * 16 + M] = (__bf16)silu_f(c[nt][v]);
  __syncthreads();

  v8f c2[8];
#pragma unroll
  for (int nt = 0; nt < 8; ++nt) c2[nt] = splat8(bn2[nt * 16 + M]);
#pragma unroll
  for (int kc = 0; kc < 4; ++kc) {
    v16bf a;
#pragma unroll
    for (int t = 0; t < 16; ++t) a[t] = hulds[wave][M * 128 + kc * 32 + kmap(t, hi)];
#pragma unroll
    for (int nt = 0; nt < 8; ++nt)
      c2[nt] = wmma_bf16(a, ldfrag(Wn2f, kc, nt, 8, lane), c2[nt]);
  }
#pragma unroll
  for (int nt = 0; nt < 8; ++nt)
#pragma unroll
    for (int v = 0; v < 8; ++v)
      h[(size_t)(m0 + v + 8 * hi) * 128 + nt * 16 + M] += c2[nt][v];
}

// ----------------------- output projection (h @ Wout) ----------------------
__global__ void __launch_bounds__(128)
eg_out(const float* __restrict__ h, const __bf16* __restrict__ Woutf,
       const float* __restrict__ bout, float* __restrict__ out, int fo) {
  const int lane = threadIdx.x & 31;
  const int wave = threadIdx.x >> 5;
  const int m0 = (blockIdx.x * 4 + wave) * 16;
  const int M = lane & 15, hi = lane >> 4;
  const int NT = fo >> 4;
  for (int nt = 0; nt < NT; ++nt) {
    v8f c = splat8(bout[nt * 16 + M]);
#pragma unroll
    for (int kc = 0; kc < 4; ++kc) {
      v16bf a;
#pragma unroll
      for (int t = 0; t < 16; ++t)
        a[t] = (__bf16)h[(size_t)(m0 + M) * 128 + kc * 32 + kmap(t, hi)];
      c = wmma_bf16(a, ldfrag(Woutf, kc, nt, NT, lane), c);
    }
#pragma unroll
    for (int v = 0; v < 8; ++v)
      out[(size_t)(m0 + v + 8 * hi) * fo + nt * 16 + M] = c[v];
  }
}

// ------------------- softmax over atoms + entropy loss ---------------------
__global__ void eg_softmax(float* __restrict__ proj, float* __restrict__ scal) {
  int t = blockIdx.x * blockDim.x + threadIdx.x;
  if (t >= BB * NBEAD) return;
  int b = t >> 4, k = t & 15;
  float* col = proj + (size_t)b * NN_ * NBEAD + k;
  float mx = -1e30f;
  for (int i = 0; i < NN_; ++i) mx = fmaxf(mx, col[i * NBEAD]);
  float sum = 0.f;
  for (int i = 0; i < NN_; ++i) sum += __expf(col[i * NBEAD] - mx);
  float inv = 1.0f / sum, ent = 0.f;
  for (int i = 0; i < NN_; ++i) {
    float p = __expf(col[i * NBEAD] - mx) * inv;
    col[i * NBEAD] = p;
    ent += -p * logf(p + 1e-15f);
  }
  atomicAdd(&scal[1], ent);
}

// ---------------- link/diag losses from ppt = proj @ proj^T ----------------
__global__ void eg_loss(const float* __restrict__ proj, const float* __restrict__ dist,
                        float* __restrict__ scal) {
  __shared__ float r1[256], r2[256];
  int t = blockIdx.x * 256 + threadIdx.x;           // t < B*N*N exactly
  int b = t >> 12, i = (t >> 6) & 63, j = t & 63;
  const float* pi = proj + ((size_t)b * NN_ + i) * NBEAD;
  const float* pj = proj + ((size_t)b * NN_ + j) * NBEAD;
  float p = 0.f;
#pragma unroll
  for (int k = 0; k < NBEAD; ++k) p += pi[k] * pj[k];
  float v = dist[t] * p;
  r1[threadIdx.x] = v * v;
  r2[threadIdx.x] = (i == j) ? fmaxf(p - 1.0f, 0.0f) : 0.0f;
  __syncthreads();
  for (int s = 128; s > 0; s >>= 1) {
    if (threadIdx.x < s) { r1[threadIdx.x] += r1[threadIdx.x + s];
                           r2[threadIdx.x] += r2[threadIdx.x + s]; }
    __syncthreads();
  }
  if (threadIdx.x == 0) { atomicAdd(&scal[0], r1[0]); atomicAdd(&scal[2], r2[0]); }
}

// -------- bead_x = proj^T @ x, centered; also seed depool coordinates ------
__global__ void eg_beadx(const float* __restrict__ proj, const float* __restrict__ x,
                         float* __restrict__ beadx, float* __restrict__ xcurb) {
  __shared__ float bx[48];
  int b = blockIdx.x, t = threadIdx.x;              // 48 threads: 16 beads x 3
  int nb = t / 3, c = t % 3;
  float s = 0.f;
  for (int i = 0; i < NN_; ++i)
    s += proj[((size_t)b * NN_ + i) * NBEAD + nb] * x[((size_t)b * NN_ + i) * 3 + c];
  bx[t] = s;
  __syncthreads();
  float mean = 0.f;
  for (int k = 0; k < NBEAD; ++k) mean += bx[k * 3 + c];
  mean *= (1.0f / NBEAD);
  float v = s - mean;
  beadx[((size_t)b * NBEAD + nb) * 3 + c] = v;
  xcurb[((size_t)b * NBEAD + nb) * 3 + c] = v;
}

// ----------------------- mean_x = inv_proj^T @ bead_x ----------------------
__global__ void eg_meanx(const float* __restrict__ ivp, const float* __restrict__ beadx,
                         float* __restrict__ out) {
  int b = blockIdx.x, t = threadIdx.x;              // 96 threads: 32 res x 3
  int r = t / 3, c = t % 3;
  float s = 0.f;
  for (int nb = 0; nb < NBEAD; ++nb)
    s += ivp[((size_t)b * NBEAD + nb) * NRES + r] * beadx[((size_t)b * NBEAD + nb) * 3 + c];
  out[((size_t)b * NRES + r) * 3 + c] = s;
}

__global__ void eg_finalize(const float* __restrict__ scal, float* __restrict__ out) {
  out[BB * NRES * 3 + 0] = sqrtf(scal[0]);
  out[BB * NRES * 3 + 1] = scal[1] * (1.0f / (BB * NBEAD));
  out[BB * NRES * 3 + 2] = scal[2];
}

// ---------------------------------------------------------------------------
extern "C" void kernel_launch(void* const* d_in, const int* in_sizes, int n_in,
                              void* d_out, int out_size, void* d_ws, size_t ws_size,
                              hipStream_t stream) {
  (void)in_sizes; (void)n_in; (void)out_size; (void)ws_size;
  const float* x      = (const float*)d_in[0];
  const int*   nfeat  = (const int*)d_in[1];
  const int*   eattr  = (const int*)d_in[3];
  const float* dist   = (const float*)d_in[4];
  const int*   bfeat  = (const int*)d_in[5];
  const int*   beattr = (const int*)d_in[7];
  const int P0 = 8, D0 = 32;
  auto LW = [&](int base, int l, int k) -> const float* {
    // k: 0 Wc 1 We1 2 We2 3 Wn1 4 Wn2 5 bc 6 be1 7 be2 8 bn1 9 bn2
    return (const float*)d_in[base + 3 + l * 10 + k];
  };

  // ---- workspace bump allocator (≈12.5 MB) ----
  char* wsb = (char*)d_ws; size_t off = 0;
  auto alloc = [&](size_t bytes) -> void* {
    off = (off + 63) & ~(size_t)63;
    void* p = wsb + off; off += bytes; return p;
  };
  float* h    = (float*)alloc((size_t)8192 * 128 * 4);
  float* agg  = (float*)alloc((size_t)8192 * 128 * 4);
  float* xc   = (float*)alloc((size_t)8192 * 3 * 4);
  float* xa   = (float*)alloc((size_t)8192 * 3 * 4);
  float* proj = (float*)alloc((size_t)BB * NN_ * NBEAD * 4);
  float* beadx= (float*)alloc((size_t)BB * NBEAD * 3 * 4);
  float* hb   = (float*)alloc((size_t)2048 * 128 * 4);
  float* aggb = (float*)alloc((size_t)2048 * 128 * 4);
  float* xcb  = (float*)alloc((size_t)2048 * 3 * 4);
  float* xab  = (float*)alloc((size_t)2048 * 3 * 4);
  float* ivp  = (float*)alloc((size_t)BB * NBEAD * NRES * 4);
  float* scal = (float*)alloc(16);
  __bf16 *We1f[2][2], *We2f[2][2], *Wn1f[2][2], *Wn2f[2][2], *Woutf[2], *eembf[2];
  for (int n = 0; n < 2; ++n)
    for (int l = 0; l < 2; ++l) {
      We1f[n][l] = (__bf16*)alloc((size_t)416 * 128 * 2);
      We2f[n][l] = (__bf16*)alloc((size_t)128 * 128 * 2);
      Wn1f[n][l] = (__bf16*)alloc((size_t)256 * 128 * 2);
      Wn2f[n][l] = (__bf16*)alloc((size_t)128 * 128 * 2);
    }
  Woutf[0] = (__bf16*)alloc((size_t)128 * 16 * 2);
  Woutf[1] = (__bf16*)alloc((size_t)128 * 32 * 2);
  eembf[0] = (__bf16*)alloc((size_t)3 * 128 * 2);
  eembf[1] = (__bf16*)alloc((size_t)3 * 128 * 2);

  // ---- prepack weights (bf16, WMMA-fragment order) ----
  auto prep = [&](const float* s, const float* bias, __bf16* d, int Ksrc, int Kpad,
                  int Nn, int mode) {
    int total = (Kpad / 32) * (Nn / 16) * 512;
    eg_prep_frag<<<(total + 255) / 256, 256, 0, stream>>>(s, bias, d, Ksrc, Kpad,
                                                          Nn, mode);
  };
  for (int n = 0; n < 2; ++n) {
    int base = n ? D0 : P0;
    for (int l = 0; l < 2; ++l) {
      prep(LW(base, l, 1), LW(base, l, 6), We1f[n][l], 385, 416, 128, 1);
      prep(LW(base, l, 2), LW(base, l, 7), We2f[n][l], 128, 128, 128, 0);
      prep(LW(base, l, 3), LW(base, l, 8), Wn1f[n][l], 256, 256, 128, 0);
      prep(LW(base, l, 4), LW(base, l, 9), Wn2f[n][l], 128, 128, 128, 0);
    }
    prep((const float*)d_in[base + 0], (const float*)d_in[base + 1], Woutf[n],
         128, 128, n ? 32 : 16, 0);
    eg_prep_bf16<<<2, 256, 0, stream>>>((const float*)d_in[base + 2], eembf[n], 384);
  }
  eg_zero_scal<<<1, 32, 0, stream>>>(scal);

  // ---- pool EGNN: 8192 nodes, n_per = 64 ----
  eg_init_h<<<4096, 256, 0, stream>>>((const float*)d_in[P0 + 23], nfeat, h, 8192);
  eg_copy_f32<<<96, 256, 0, stream>>>(x, xc, 8192 * 3);
  for (int l = 0; l < 2; ++l) {
    eg_edge<<<512, 128, E_LDS_BYTES, stream>>>(h, xc, eembf[0], eattr,
        We1f[0][l], We2f[0][l], LW(P0, l, 7),
        LW(P0, l, 0), LW(P0, l, 5), agg, xa, 64);
    eg_node<<<128, 128, 0, stream>>>(h, agg, Wn1f[0][l], LW(P0, l, 8),
                                     Wn2f[0][l], LW(P0, l, 9));
    eg_xupd<<<96, 256, 0, stream>>>(xc, xa, 1.0f / 64.0f, 8192 * 3);
  }
  eg_out<<<128, 128, 0, stream>>>(h, Woutf[0], (const float*)d_in[P0 + 1], proj, 16);
  eg_softmax<<<8, 256, 0, stream>>>(proj, scal);
  eg_loss<<<2048, 256, 0, stream>>>(proj, dist, scal);
  eg_beadx<<<128, 48, 0, stream>>>(proj, x, beadx, xcb);

  // ---- depool EGNN: 2048 bead nodes, n_per = 16 ----
  eg_init_h<<<1024, 256, 0, stream>>>((const float*)d_in[D0 + 23], bfeat, hb, 2048);
  for (int l = 0; l < 2; ++l) {
    eg_edge<<<128, 128, E_LDS_BYTES, stream>>>(hb, xcb, eembf[1], beattr,
        We1f[1][l], We2f[1][l], LW(D0, l, 7),
        LW(D0, l, 0), LW(D0, l, 5), aggb, xab, 16);
    eg_node<<<32, 128, 0, stream>>>(hb, aggb, Wn1f[1][l], LW(D0, l, 8),
                                    Wn2f[1][l], LW(D0, l, 9));
    eg_xupd<<<24, 256, 0, stream>>>(xcb, xab, 1.0f / 16.0f, 2048 * 3);
  }
  eg_out<<<32, 128, 0, stream>>>(hb, Woutf[1], (const float*)d_in[D0 + 1], ivp, 32);
  eg_meanx<<<128, 96, 0, stream>>>(ivp, beadx, (float*)d_out);
  eg_finalize<<<1, 1, 0, stream>>>(scal, (float*)d_out);
}
